// CycleNet_61838939128046
// MI455X (gfx1250) — matrix operand order repaired and checked by reading, not compiled, not run on hardware
//
#include <hip/hip_runtime.h>
#include <hip/hip_bf16.h>
#include <stdint.h>

// ---------------- problem constants (match reference) ----------------
#define Hd   128
#define Nn   100000
#define Ee   250000
#define N5c  20000
#define N6c  30000
#define Gg   512
#define Ll   3
#define AFf  9
#define AVv  64
#define BFf  3
#define BVv  8

#define NC5  (N5c * 5)      // 100000 cycle positions (5-cycles)
#define NC6  (N6c * 6)      // 180000 cycle positions (6-cycles)

typedef __attribute__((ext_vector_type(16))) __bf16         v16bf;
typedef __attribute__((ext_vector_type(16))) unsigned short v16us;
typedef __attribute__((ext_vector_type(8)))  unsigned short v8us;
typedef __attribute__((ext_vector_type(8)))  float          v8f;

// ---------------- generic bf16-WMMA GEMM ----------------
// C[M,Ncols] = epilogue(A[gather,K](bf16) @ B[K,Ncols](f32 -> bf16 LDS))
// ep: 0 -> C = acc + bias
//     1 -> C += acc
//     2 -> C = C + relu(acc + bias)   (residual in place; also writes Cbf if set)
//     3 -> C = relu(acc + bias)
// One 16x16 output tile per wave, 8 waves / 256-thread block.
// A is pre-converted bf16 (producers are memory-bound elementwise passes that
// convert for free) -> inner loop is 2x global b128 + 2x ds b128 + 1x wmma.
template <int K>
__global__ __launch_bounds__(256) void gemm_bf16_wmma(
    const __bf16* __restrict__ A, const int* __restrict__ rowidx,
    const float* __restrict__ B, const float* __restrict__ bias,
    float* __restrict__ C, __bf16* __restrict__ Cbf, int M, int Ncols, int ep) {
  __shared__ __align__(32) unsigned short sB[K * 16];  // sB[col*K + k], bf16 bits
  const int tid = threadIdx.x;
  const int n0  = blockIdx.y * 16;

  for (int idx = tid; idx < K * 16; idx += 256) {
    int col = idx / K, k = idx - col * K;
    __bf16 b = (__bf16)B[(size_t)k * Ncols + n0 + col];
    sB[idx] = __builtin_bit_cast(unsigned short, b);
  }
  __syncthreads();

  const int wid    = tid >> 5;
  const int lane   = tid & 31;
  const int tile_m = blockIdx.x * 8 + wid;
  if (tile_m * 16 < M) {
    const int hi = lane >> 4;   // half-wave select
    const int lo = lane & 15;

    int row16 = tile_m * 16 + lo;
    int ar    = rowidx ? rowidx[row16] : row16;
    const __bf16* arow = A + (size_t)ar * K;

    v8f acc = {};
#pragma unroll
    for (int k0 = 0; k0 < K; k0 += 32) {
      // A fragment: lane holds K runs [k0+8*hi, +8) and [k0+16+8*hi, +8)
      v8us a0 = *(const v8us*)(arow + k0 + 8 * hi);        // 16B global load
      v8us a1 = *(const v8us*)(arow + k0 + 16 + 8 * hi);   // 16B global load
      v16us au = __builtin_shufflevector(a0, a1, 0, 1, 2, 3, 4, 5, 6, 7,
                                         8, 9, 10, 11, 12, 13, 14, 15);
      v16bf a = __builtin_bit_cast(v16bf, au);

      // B fragment: lane (col=lo) holds K runs [k0+8*hi, +8) and [k0+16+8*hi, +8)
      v8us b0 = *(const v8us*)&sB[lo * K + k0 + 8 * hi];
      v8us b1 = *(const v8us*)&sB[lo * K + k0 + 16 + 8 * hi];
      v16us bu = __builtin_shufflevector(b0, b1, 0, 1, 2, 3, 4, 5, 6, 7,
                                         8, 9, 10, 11, 12, 13, 14, 15);
      v16bf b  = __builtin_bit_cast(v16bf, bu);

      acc = __builtin_amdgcn_wmma_f32_16x16x32_bf16(
          false, a, false, b, (short)0, acc, false, false);
    }

    const int   col   = n0 + lo;
    const float bs    = bias ? bias[col] : 0.0f;
    const int   rbase = tile_m * 16 + 8 * hi;
#pragma unroll
    for (int r = 0; r < 8; ++r) {
      float  v  = acc[r];
      size_t ci = (size_t)(rbase + r) * Ncols + col;
      float* cp = C + ci;
      if (ep == 0)      *cp = v + bs;
      else if (ep == 1) *cp = *cp + v;
      else if (ep == 2) {
        float nv = *cp + fmaxf(v + bs, 0.0f);
        *cp = nv;
        if (Cbf) Cbf[ci] = (__bf16)nv;   // bf16 shadow for downstream GEMM A
      } else            *cp = fmaxf(v + bs, 0.0f);
    }
  }
}

// ---------------- elementwise / scatter helpers ----------------
__global__ void zero_f32_k(float* __restrict__ p, size_t n) {
  size_t i = (size_t)blockIdx.x * blockDim.x + threadIdx.x;
  size_t stride = (size_t)gridDim.x * blockDim.x;
  for (; i < n; i += stride) p[i] = 0.0f;
}

__global__ void cvt_f32_bf16_k(const float* __restrict__ src, __bf16* __restrict__ dst,
                               int total) {
  int i = blockIdx.x * blockDim.x + threadIdx.x;
  if (i >= total) return;
  dst[i] = (__bf16)src[i];
}

__global__ void atom_enc_k(const int* __restrict__ xa, const float* __restrict__ aemb,
                           float* __restrict__ X) {
  int gid = blockIdx.x * blockDim.x + threadIdx.x;
  if (gid >= Nn * Hd) return;
  int n = gid >> 7, hh = gid & 127;
  float s = 0.0f;
#pragma unroll
  for (int f = 0; f < AFf; ++f) {
    int a = xa[(size_t)n * AFf + f];
    s += aemb[((size_t)f * AVv + a) * Hd + hh];
  }
  X[gid] = s;
}

__global__ void cyc_init_k(const int* __restrict__ xc, const float* __restrict__ cemb,
                           float* __restrict__ out, int total) {
  int gid = blockIdx.x * blockDim.x + threadIdx.x;
  if (gid >= total) return;
  out[gid] = cemb[(size_t)xc[gid >> 7] * Hd + (gid & 127)];
}

__global__ void build_idx_k(int* __restrict__ prev, int* __restrict__ next, int n, int k) {
  int gid = blockIdx.x * blockDim.x + threadIdx.x;
  if (gid >= n * k) return;
  int c = gid / k, j = gid - c * k;
  prev[gid] = c * k + (j + k - 1) % k;
  next[gid] = c * k + (j + 1) % k;
}

__global__ void init_h_k(const float* __restrict__ X, const float* __restrict__ eps,
                         float* __restrict__ Hb, int total) {
  int gid = blockIdx.x * blockDim.x + threadIdx.x;
  if (gid >= total) return;
  Hb[gid] = (1.0f + eps[0]) * X[gid];
}

__global__ __launch_bounds__(256) void gine_agg_k(
    const float* __restrict__ X, const float* __restrict__ bembL,
    const int* __restrict__ eattr, const int* __restrict__ src,
    const int* __restrict__ dst, float* __restrict__ Hb) {
  int gid = blockIdx.x * blockDim.x + threadIdx.x;
  if (gid >= Ee * 32) return;
  int e = gid >> 5, q = gid & 31;
  int s = src[e], d = dst[e];
  float4 xv = ((const float4*)(X + (size_t)s * Hd))[q];
  float ex = 0.f, ey = 0.f, ez = 0.f, ew = 0.f;
#pragma unroll
  for (int f = 0; f < BFf; ++f) {
    int a = eattr[(size_t)e * BFf + f];
    float4 t = ((const float4*)(bembL + ((size_t)f * BVv + a) * Hd))[q];
    ex += t.x; ey += t.y; ez += t.z; ew += t.w;
  }
  float* o = Hb + (size_t)d * Hd + q * 4;
  atomicAdd(o + 0, fmaxf(xv.x + ex, 0.0f));
  atomicAdd(o + 1, fmaxf(xv.y + ey, 0.0f));
  atomicAdd(o + 2, fmaxf(xv.z + ez, 0.0f));
  atomicAdd(o + 3, fmaxf(xv.w + ew, 0.0f));
}

__global__ __launch_bounds__(256) void bn_stats_k(const float* __restrict__ Xm, int M, int C,
                                                  float* __restrict__ stat) {
  int col = threadIdx.x;
  if (col >= C) return;
  int r0 = blockIdx.x * 256;
  int r1 = r0 + 256 < M ? r0 + 256 : M;
  float s = 0.0f, ss = 0.0f;
  for (int r = r0; r < r1; ++r) {
    float v = Xm[(size_t)r * C + col];
    s += v; ss += v * v;
  }
  atomicAdd(&stat[col], s);
  atomicAdd(&stat[C + col], ss);
}

__global__ void bn_fin_k(const float* __restrict__ stat, const float* __restrict__ g,
                         const float* __restrict__ b, int M, int C, float* __restrict__ scsh) {
  int c = blockIdx.x * blockDim.x + threadIdx.x;
  if (c >= C) return;
  float inv  = 1.0f / (float)M;
  float mean = stat[c] * inv;
  float var  = stat[C + c] * inv - mean * mean;
  float sc   = g[c] * rsqrtf(var + 1e-5f);
  scsh[c]     = sc;
  scsh[C + c] = b[c] - mean * sc;
}

// BN scale/shift + relu; writes f32 and/or bf16 (either may be null).
__global__ void bn_apply_relu_k(const float* __restrict__ src, float* __restrict__ dstf,
                                __bf16* __restrict__ dstb, const float* __restrict__ scsh,
                                int total, int C) {
  int i = blockIdx.x * blockDim.x + threadIdx.x;
  if (i >= total) return;
  int c = i % C;
  float v = fmaxf(fmaf(src[i], scsh[c], scsh[C + c]), 0.0f);
  if (dstf) dstf[i] = v;
  if (dstb) dstb[i] = (__bf16)v;
}

__global__ void add_relu_k(float* __restrict__ dst, const float* __restrict__ src, int n) {
  int i = blockIdx.x * blockDim.x + threadIdx.x;
  if (i >= n) return;
  dst[i] += fmaxf(src[i], 0.0f);
}

__global__ __launch_bounds__(256) void scatter_sum_k(
    const float* __restrict__ vals, const int* __restrict__ idx,
    float* __restrict__ sums, float* __restrict__ cnt, int M) {
  int gid = blockIdx.x * blockDim.x + threadIdx.x;
  if (gid >= M * 32) return;
  int r = gid >> 5, q = gid & 31;
  int t = idx[r];
  float4 v = ((const float4*)(vals + (size_t)r * Hd))[q];
  float* o = sums + (size_t)t * Hd + q * 4;
  atomicAdd(o + 0, v.x); atomicAdd(o + 1, v.y);
  atomicAdd(o + 2, v.z); atomicAdd(o + 3, v.w);
  if (q == 0) atomicAdd(&cnt[t], 1.0f);
}

// sums(f32) / count -> bf16 (feeds next GEMM's A operand)
__global__ void seg_div_bf_k(const float* __restrict__ sums, const float* __restrict__ cnt,
                             __bf16* __restrict__ dstb, int total) {
  int i = blockIdx.x * blockDim.x + threadIdx.x;
  if (i >= total) return;
  dstb[i] = (__bf16)(sums[i] / fmaxf(cnt[i >> 7], 1.0f));
}

__global__ void final_out_k(const float* __restrict__ xg2, const float* __restrict__ lw,
                            const float* __restrict__ lb, float* __restrict__ out) {
  int g = blockIdx.x * blockDim.x + threadIdx.x;
  if (g >= Gg) return;
  float acc = lb[0];
  const float* r = xg2 + (size_t)g * Hd;
#pragma unroll
  for (int t = 0; t < Hd; ++t) acc += r[t] * lw[t];
  out[g] = acc;
}

// ---------------- host-side orchestration ----------------
static inline int cdiv(long long a, int b) { return (int)((a + b - 1) / b); }

static void gemm(const __bf16* A, const int* ridx, const float* B, const float* bias,
                 float* C, __bf16* Cbf, int M, int K, int Ncols, int ep, hipStream_t s) {
  dim3 grid(cdiv(M / 16, 8), Ncols / 16);
  if (K == 128)
    gemm_bf16_wmma<128><<<grid, 256, 0, s>>>(A, ridx, B, bias, C, Cbf, M, Ncols, ep);
  else
    gemm_bf16_wmma<256><<<grid, 256, 0, s>>>(A, ridx, B, bias, C, Cbf, M, Ncols, ep);
}

static void zero(float* p, size_t n, hipStream_t s) {
  zero_f32_k<<<1024, 256, 0, s>>>(p, n);
}

extern "C" void kernel_launch(void* const* d_in, const int* in_sizes, int n_in,
                              void* d_out, int out_size, void* d_ws, size_t ws_size,
                              hipStream_t stream) {
  const int*   x_atom     = (const int*)d_in[0];
  const int*   edge_index = (const int*)d_in[1];
  const int*   edge_attr  = (const int*)d_in[2];
  const int*   batch      = (const int*)d_in[3];
  const int*   xc5        = (const int*)d_in[4];
  const int*   xc6        = (const int*)d_in[5];
  const int*   a2c5_row   = (const int*)d_in[6];
  const int*   a2c6_row   = (const int*)d_in[8];
  const float* atom_emb   = (const float*)d_in[10];
  const float* bond_emb   = (const float*)d_in[11];
  const float* cyc_emb5   = (const float*)d_in[12];
  const float* cyc_emb6   = (const float*)d_in[13];
  const float* gine_eps   = (const float*)d_in[14];
  const float* gw1 = (const float*)d_in[15];
  const float* gb1 = (const float*)d_in[16];
  const float* gbn_g = (const float*)d_in[17];
  const float* gbn_b = (const float*)d_in[18];
  const float* gw2 = (const float*)d_in[19];
  const float* gb2 = (const float*)d_in[20];
  const float* bn_g = (const float*)d_in[21];
  const float* bn_b = (const float*)d_in[22];
  const float* a2c5_w = (const float*)d_in[23];
  const float* a2c5_b = (const float*)d_in[24];
  const float* a2c6_w = (const float*)d_in[25];
  const float* a2c6_b = (const float*)d_in[26];
  const float* c2a5_w = (const float*)d_in[27];
  const float* c2a5_b = (const float*)d_in[28];
  const float* c2a6_w = (const float*)d_in[29];
  const float* c2a6_b = (const float*)d_in[30];
  const float* p5_w = (const float*)d_in[31];
  const float* p5_b = (const float*)d_in[32];
  const float* p6_w = (const float*)d_in[33];
  const float* p6_b = (const float*)d_in[34];
  const float* atom_lin_w = (const float*)d_in[35];
  const float* atom_lin_b = (const float*)d_in[36];
  const float* lin_w = (const float*)d_in[37];
  const float* lin_b = (const float*)d_in[38];
  float* out = (float*)d_out;

  const int* esrc = edge_index;
  const int* edst = edge_index + Ee;

  // ---- workspace carve-up (units: floats / 4 bytes) ----
  float* W = (float*)d_ws;
  size_t o = 0;
  float* X    = W + o; o += (size_t)Nn * Hd;        // atom features (f32)
  float* X5   = W + o; o += (size_t)NC5 * Hd;       // 5-cycle features (f32)
  float* X6   = W + o; o += (size_t)NC6 * Hd;       // 6-cycle features (f32)
  float* HB   = W + o; o += (size_t)Nn * Hd;        // h / h2 (f32)
  float* BIG  = W + o; o += (size_t)Nn * 2 * Hd;    // h1 / conv tmp / seg-mean sums
  float* CNT  = W + o; o += (size_t)Nn;             // seg-mean counts
  float* STAT = W + o; o += 512;                    // BN sums/sumsq
  float* SCSH = W + o; o += 512;                    // BN scale/shift
  float* XG2  = W + o; o += (size_t)Gg * Hd;        // readout hidden
  int* prev5 = (int*)(W + o); o += NC5;
  int* next5 = (int*)(W + o); o += NC5;
  int* prev6 = (int*)(W + o); o += NC6;
  int* next6 = (int*)(W + o); o += NC6;
  // bf16 GEMM A-operand buffers (2 bytes/elem; sizes below in float units)
  __bf16* ABUF = (__bf16*)(W + o); o += (size_t)Nn * Hd;       // shared: h / h1relu / seg-means (<= N*2H bf16)
  __bf16* XH   = (__bf16*)(W + o); o += (size_t)Nn * Hd / 2;   // x snapshot (a2c input)
  __bf16* X5H  = (__bf16*)(W + o); o += (size_t)NC5 * Hd / 2;  // x5 snapshot (path input)
  __bf16* X6H  = (__bf16*)(W + o); o += (size_t)NC6 * Hd / 2;  // x6 snapshot (path input)
  (void)ws_size; (void)in_sizes; (void)n_in; (void)out_size;

  // ---- encoders + cycle index tables ----
  atom_enc_k<<<cdiv((long long)Nn * Hd, 256), 256, 0, stream>>>(x_atom, atom_emb, X);
  cyc_init_k<<<cdiv((long long)NC5 * Hd, 256), 256, 0, stream>>>(xc5, cyc_emb5, X5, NC5 * Hd);
  cyc_init_k<<<cdiv((long long)NC6 * Hd, 256), 256, 0, stream>>>(xc6, cyc_emb6, X6, NC6 * Hd);
  build_idx_k<<<cdiv(NC5, 256), 256, 0, stream>>>(prev5, next5, N5c, 5);
  build_idx_k<<<cdiv(NC6, 256), 256, 0, stream>>>(prev6, next6, N6c, 6);

  for (int i = 0; i < Ll; ++i) {
    // -------- GINEConv: h = (1+eps)x + sum relu(x[src]+ea) --------
    init_h_k<<<cdiv((long long)Nn * Hd, 256), 256, 0, stream>>>(X, gine_eps + i, HB, Nn * Hd);
    gine_agg_k<<<cdiv((long long)Ee * 32, 256), 256, 0, stream>>>(
        X, bond_emb + (size_t)i * BFf * BVv * Hd, edge_attr, esrc, edst, HB);
    cvt_f32_bf16_k<<<cdiv((long long)Nn * Hd, 256), 256, 0, stream>>>(HB, ABUF, Nn * Hd);

    // -------- MLP: relu(BN(h@gw1+gb1)) @ gw2 + gb2, then x = relu(BN(.)) --------
    gemm(ABUF, nullptr, gw1 + (size_t)i * Hd * 2 * Hd, gb1 + (size_t)i * 2 * Hd,
         BIG, nullptr, Nn, Hd, 2 * Hd, 0, stream);
    zero(STAT, 512, stream);
    bn_stats_k<<<cdiv(Nn, 256), 256, 0, stream>>>(BIG, Nn, 2 * Hd, STAT);
    bn_fin_k<<<1, 256, 0, stream>>>(STAT, gbn_g + (size_t)i * 2 * Hd, gbn_b + (size_t)i * 2 * Hd,
                                    Nn, 2 * Hd, SCSH);
    bn_apply_relu_k<<<cdiv((long long)Nn * 2 * Hd, 256), 256, 0, stream>>>(
        BIG, nullptr, ABUF, SCSH, Nn * 2 * Hd, 2 * Hd);  // bf16 only: feeds gemm2
    gemm(ABUF, nullptr, gw2 + (size_t)i * 2 * Hd * Hd, gb2 + (size_t)i * Hd,
         HB, nullptr, Nn, 2 * Hd, Hd, 0, stream);
    zero(STAT, 512, stream);
    bn_stats_k<<<cdiv(Nn, 256), 256, 0, stream>>>(HB, Nn, Hd, STAT);
    bn_fin_k<<<1, 256, 0, stream>>>(STAT, bn_g + (size_t)i * Hd, bn_b + (size_t)i * Hd,
                                    Nn, Hd, SCSH);
    bn_apply_relu_k<<<cdiv((long long)Nn * Hd, 256), 256, 0, stream>>>(
        HB, X, XH, SCSH, Nn * Hd, Hd);                   // f32 X + bf16 snapshot

    // -------- atoms -> cycles (identity col => pure row-gather GEMM, residual relu) ----
    gemm(XH, a2c5_row, a2c5_w + (size_t)i * Hd * Hd, a2c5_b + (size_t)i * Hd,
         X5, X5H, NC5, Hd, Hd, 2, stream);               // writes X5 + bf16 snapshot
    gemm(XH, a2c6_row, a2c6_w + (size_t)i * Hd * Hd, a2c6_b + (size_t)i * Hd,
         X6, X6H, NC6, Hd, Hd, 2, stream);

    // -------- cyclic path blocks: x += relu(x_prev@W0 + x@W1 + x_next@W2 + b) --------
    {
      const float* W5 = p5_w + (size_t)i * 3 * Hd * Hd;
      gemm(X5H, prev5,   W5 + 0 * Hd * Hd, p5_b + (size_t)i * Hd, BIG, nullptr, NC5, Hd, Hd, 0, stream);
      gemm(X5H, nullptr, W5 + 1 * Hd * Hd, nullptr,               BIG, nullptr, NC5, Hd, Hd, 1, stream);
      gemm(X5H, next5,   W5 + 2 * Hd * Hd, nullptr,               BIG, nullptr, NC5, Hd, Hd, 1, stream);
      add_relu_k<<<cdiv((long long)NC5 * Hd, 256), 256, 0, stream>>>(X5, BIG, NC5 * Hd);

      const float* W6 = p6_w + (size_t)i * 3 * Hd * Hd;
      gemm(X6H, prev6,   W6 + 0 * Hd * Hd, p6_b + (size_t)i * Hd, BIG, nullptr, NC6, Hd, Hd, 0, stream);
      gemm(X6H, nullptr, W6 + 1 * Hd * Hd, nullptr,               BIG, nullptr, NC6, Hd, Hd, 1, stream);
      gemm(X6H, next6,   W6 + 2 * Hd * Hd, nullptr,               BIG, nullptr, NC6, Hd, Hd, 1, stream);
      add_relu_k<<<cdiv((long long)NC6 * Hd, 256), 256, 0, stream>>>(X6, BIG, NC6 * Hd);
    }

    // -------- cycles -> atoms: x += relu(seg_mean(x5, row) @ W + b) --------
    zero(BIG, (size_t)Nn * Hd, stream);
    zero(CNT, (size_t)Nn, stream);
    scatter_sum_k<<<cdiv((long long)NC5 * 32, 256), 256, 0, stream>>>(X5, a2c5_row, BIG, CNT, NC5);
    seg_div_bf_k<<<cdiv((long long)Nn * Hd, 256), 256, 0, stream>>>(BIG, CNT, ABUF, Nn * Hd);
    gemm(ABUF, nullptr, c2a5_w + (size_t)i * Hd * Hd, c2a5_b + (size_t)i * Hd,
         X, nullptr, Nn, Hd, Hd, 2, stream);

    zero(BIG, (size_t)Nn * Hd, stream);
    zero(CNT, (size_t)Nn, stream);
    scatter_sum_k<<<cdiv((long long)NC6 * 32, 256), 256, 0, stream>>>(X6, a2c6_row, BIG, CNT, NC6);
    seg_div_bf_k<<<cdiv((long long)Nn * Hd, 256), 256, 0, stream>>>(BIG, CNT, ABUF, Nn * Hd);
    gemm(ABUF, nullptr, c2a6_w + (size_t)i * Hd * Hd, c2a6_b + (size_t)i * Hd,
         X, nullptr, Nn, Hd, Hd, 2, stream);
  }

  // -------- graph readout --------
  zero(BIG, (size_t)Gg * Hd, stream);
  zero(CNT, (size_t)Gg, stream);
  scatter_sum_k<<<cdiv((long long)Nn * 32, 256), 256, 0, stream>>>(X, batch, BIG, CNT, Nn);
  seg_div_bf_k<<<cdiv((long long)Gg * Hd, 256), 256, 0, stream>>>(BIG, CNT, ABUF, Gg * Hd);
  gemm(ABUF, nullptr, atom_lin_w, atom_lin_b, XG2, nullptr, Gg, Hd, Hd, 3, stream);
  final_out_k<<<cdiv(Gg, 256), 256, 0, stream>>>(XG2, lin_w, lin_b, out);
}
